// DifcannyLoss_11802570130032
// MI455X (gfx1250) — matrix operand order repaired and checked by reading, not compiled, not run on hardware
//
#include <hip/hip_runtime.h>
#include <hip/hip_bf16.h>

typedef __attribute__((ext_vector_type(16))) _Float16 v16h;
typedef __attribute__((ext_vector_type(8)))  float    v8f;

#define NIMG 16
#define HH   1024
#define WW   1024
#define HW   (HH * WW)          // 1048576 = 2^20
#define NHW  (NIMG * HW)        // 16777216

// Normalized 17-tap Gaussian, sigma=2.0, r=8 (matches reference init).
__device__ __constant__ float G17[17] = {
    6.69192e-05f, 4.36345e-04f, 2.21596e-03f, 8.76431e-03f, 2.69960e-02f,
    6.47599e-02f, 1.20988e-01f, 1.76036e-01f, 1.99475e-01f, 1.76036e-01f,
    1.20988e-01f, 6.47599e-02f, 2.69960e-02f, 8.76431e-03f, 2.21596e-03f,
    4.36345e-04f, 6.69192e-05f};

__device__ __forceinline__ int refl(int i, int n) {
    if (i < 0) i = -i;
    if (i >= n) i = 2 * n - 2 - i;
    return i;
}

// ---------------------------------------------------------------------------
// Horizontal 17-tap blur via WMMA. One wave computes one 16x16 output tile:
//   D(16x16) = A(16x32 image slab, 8-px halo each side) x B(32x16 banded G)
// A layout (16-bit, 16x32): lane l<16 -> M=l, elems 0..7 = K 0..7, 8..15 = K 16..23
//                           lane l>=16 -> same M, K ranges +8 (K 8..15 / 24..31)
// B layout (32x16): lanes 0..15 hold K=0..7,16..23 for N=lane; lanes 16..31 hold K=8..15,24..31
// D layout (f32): VGPR v -> row v (lanes 0-15) / row v+8 (lanes 16-31), N=lane&15
// ---------------------------------------------------------------------------
__global__ __launch_bounds__(256) void blur_h_wmma(const float* __restrict__ x,
                                                   float* __restrict__ out) {
    const int lane = threadIdx.x & 31;
    const int tile = blockIdx.x * 8 + (threadIdx.x >> 5);
    const int n    = tile >> 12;            // tile / 4096
    const int rem  = tile & 4095;
    const int row0 = (rem >> 6) << 4;
    const int col0 = (rem & 63) << 4;
    const float* img = x + ((size_t)n << 20);
    float* o         = out + ((size_t)n << 20);

    const int half = lane >> 4;             // 0 or 1
    const int m16  = lane & 15;

    // B: banded Gaussian matrix, G[k][c] = g17[k-c] for 0 <= k-c <= 16
    v16h b;
#pragma unroll
    for (int e = 0; e < 16; ++e) {
        int K = half * 8 + e + ((e >= 8) ? 8 : 0);
        int d = K - m16;
        b[e] = (d >= 0 && d <= 16) ? (_Float16)G17[d] : (_Float16)0.0f;
    }

    // A: image row slab with reflect padding on columns
    v16h a;
    const float* rp = img + (size_t)(row0 + m16) * WW;
#pragma unroll
    for (int e = 0; e < 16; ++e) {
        int K = half * 8 + e + ((e >= 8) ? 8 : 0);
        int c = refl(col0 - 8 + K, WW);
        a[e] = (_Float16)rp[c];
    }

    v8f acc = {};
    acc = __builtin_amdgcn_wmma_f32_16x16x32_f16(false, a, false, b,
                                                 (short)0, acc, false, false);
#pragma unroll
    for (int v = 0; v < 8; ++v) {
        int r = row0 + half * 8 + v;
        o[(size_t)r * WW + col0 + m16] = acc[v];
    }
}

// Vertical pass: roles swapped — A holds the banded Gaussian (A[r][k]=g[k-r]),
// B holds the image column slab (B[k][c] = in[row0-8+k][col0+c]).
__global__ __launch_bounds__(256) void blur_v_wmma(const float* __restrict__ x,
                                                   float* __restrict__ out) {
    const int lane = threadIdx.x & 31;
    const int tile = blockIdx.x * 8 + (threadIdx.x >> 5);
    const int n    = tile >> 12;
    const int rem  = tile & 4095;
    const int row0 = (rem >> 6) << 4;
    const int col0 = (rem & 63) << 4;
    const float* img = x + ((size_t)n << 20);
    float* o         = out + ((size_t)n << 20);

    const int half = lane >> 4;
    const int m16  = lane & 15;

    v16h a;
#pragma unroll
    for (int e = 0; e < 16; ++e) {
        int K = half * 8 + e + ((e >= 8) ? 8 : 0);
        int d = K - m16;
        a[e] = (d >= 0 && d <= 16) ? (_Float16)G17[d] : (_Float16)0.0f;
    }

    v16h b;
#pragma unroll
    for (int e = 0; e < 16; ++e) {
        int K = half * 8 + e + ((e >= 8) ? 8 : 0);
        int r = refl(row0 - 8 + K, HH);
        b[e] = (_Float16)img[(size_t)r * WW + col0 + m16];
    }

    v8f acc = {};
    acc = __builtin_amdgcn_wmma_f32_16x16x32_f16(false, a, false, b,
                                                 (short)0, acc, false, false);
#pragma unroll
    for (int v = 0; v < 8; ++v) {
        int r = row0 + half * 8 + v;
        o[(size_t)r * WW + col0 + m16] = acc[v];
    }
}

// ---------------------------------------------------------------------------
// Sobel gradient magnitude (reflect-padded), one thread per pixel.
// ---------------------------------------------------------------------------
__global__ __launch_bounds__(256) void mag_k(const float* __restrict__ sm,
                                             float* __restrict__ mag) {
    int p = blockIdx.x * 256 + threadIdx.x;
    int n = p >> 20;
    int hw = p & (HW - 1);
    int r = hw >> 10, c = hw & 1023;
    const float* im = sm + ((size_t)n << 20);
    auto at = [&](int rr, int cc) -> float {
        return im[(size_t)refl(rr, HH) * WW + refl(cc, WW)];
    };
    float tl = at(r - 1, c - 1), tm = at(r - 1, c), tr = at(r - 1, c + 1);
    float ml = at(r, c - 1), mr = at(r, c + 1);
    float bl = at(r + 1, c - 1), bm = at(r + 1, c), br = at(r + 1, c + 1);
    float gx = (tr + 2.0f * mr + br) - (tl + 2.0f * ml + bl);
    float gy = (bl + 2.0f * bm + br) - (tl + 2.0f * tm + tr);
    mag[p] = sqrtf(gx * gx + gy * gy);
}

// ---------------------------------------------------------------------------
// Non-max suppression + double threshold -> packed strong/weak bitmaps.
// Angle bins via slope comparisons (no atan2): |gy|<tan22.5*|gx| -> E/W,
// |gy|>tan67.5*|gx| -> N/S, else diagonal picked by sign(gx*gy).
// Neighbor magnitudes are ZERO padded (matches reference's jnp.pad default).
// ---------------------------------------------------------------------------
__global__ __launch_bounds__(256) void nms_k(const float* __restrict__ sm,
                                             const float* __restrict__ mag,
                                             unsigned* __restrict__ strongBM,
                                             unsigned* __restrict__ weakBM) {
    int p = blockIdx.x * 256 + threadIdx.x;
    int n = p >> 20;
    int hw = p & (HW - 1);
    int r = hw >> 10, c = hw & 1023;
    const float* im = sm + ((size_t)n << 20);
    const float* mg = mag + ((size_t)n << 20);
    auto at = [&](int rr, int cc) -> float {
        return im[(size_t)refl(rr, HH) * WW + refl(cc, WW)];
    };
    auto mz = [&](int rr, int cc) -> float {
        return (rr < 0 || rr >= HH || cc < 0 || cc >= WW)
                   ? 0.0f
                   : mg[(size_t)rr * WW + cc];
    };
    float tl = at(r - 1, c - 1), tm = at(r - 1, c), tr = at(r - 1, c + 1);
    float ml = at(r, c - 1), mr = at(r, c + 1);
    float bl = at(r + 1, c - 1), bm = at(r + 1, c), br = at(r + 1, c + 1);
    float gx = (tr + 2.0f * mr + br) - (tl + 2.0f * ml + bl);
    float gy = (bl + 2.0f * bm + br) - (tl + 2.0f * tm + tr);
    float m = mg[(size_t)r * WW + c];

    float ax = fabsf(gx), ay = fabsf(gy);
    float n1, n2;
    if (ay < 0.41421356f * ax) {            // ~horizontal gradient
        n1 = mz(r, c + 1);  n2 = mz(r, c - 1);
    } else if (ay > 2.41421356f * ax) {     // ~vertical gradient
        n1 = mz(r - 1, c);  n2 = mz(r + 1, c);
    } else if (gx * gy > 0.0f) {            // 45 deg
        n1 = mz(r - 1, c + 1); n2 = mz(r + 1, c - 1);
    } else {                                 // 135 deg
        n1 = mz(r - 1, c - 1); n2 = mz(r + 1, c + 1);
    }
    bool keep = (m >= n1) && (m >= n2);
    bool strong = keep && (m > 0.2f);
    bool weak = keep && (m > 0.1f);

    unsigned ms = (unsigned)__ballot(strong);
    unsigned mw = (unsigned)__ballot(weak);
    if ((threadIdx.x & 31) == 0) {
        strongBM[p >> 5] = ms;
        weakBM[p >> 5]   = mw;
    }
}

// ---------------------------------------------------------------------------
// Hysteresis: one workgroup per image, strong+weak bitmaps (2x128KB) resident
// in LDS (fits the 320KB WGP LDS). Staging uses gfx1250 async cache->LDS
// copies (GLOBAL_LOAD_ASYNC_TO_LDS_B128 / GLOBAL_STORE_ASYNC_FROM_LDS_B128,
// tracked by ASYNCcnt) — no VGPR round trip. Bit-parallel 3x3 dilation,
// in-place monotone iteration to the same least fixpoint as the reference.
// Generic LDS pointers carry the LDS byte offset in their low 32 bits
// (ISA 10.2 aperture rules), which is exactly what the vdst operand wants.
// ---------------------------------------------------------------------------
#define WORDS_PER_IMG 32768     // 1024*1024/32, 32 words per row
extern __shared__ unsigned hlds[];

__device__ __forceinline__ void async_g2l_b128(void* lds_dst, const void* gsrc) {
    unsigned lds_addr = (unsigned)(uintptr_t)lds_dst;
    unsigned long long ga = (unsigned long long)(uintptr_t)gsrc;
    asm volatile("global_load_async_to_lds_b128 %0, %1, off"
                 :: "v"(lds_addr), "v"(ga) : "memory");
}
__device__ __forceinline__ void async_l2g_b128(void* gdst, const void* lds_src) {
    unsigned lds_addr = (unsigned)(uintptr_t)lds_src;
    unsigned long long ga = (unsigned long long)(uintptr_t)gdst;
    asm volatile("global_store_async_from_lds_b128 %0, %1, off"
                 :: "v"(ga), "v"(lds_addr) : "memory");
}
__device__ __forceinline__ void wait_asynccnt0() {
    asm volatile("s_wait_asynccnt 0x0" ::: "memory");
}

__global__ __launch_bounds__(1024) void hyst_k(unsigned* __restrict__ strongBM,
                                               const unsigned* __restrict__ weakBM) {
    unsigned* S    = hlds;
    unsigned* Wk   = hlds + WORDS_PER_IMG;
    unsigned* flag = hlds + 2 * WORDS_PER_IMG;
    const int tid = threadIdx.x;
    const size_t base = (size_t)blockIdx.x * WORDS_PER_IMG;

    // Async bulk stage: each lane moves 16B per issue, 8 sweeps per buffer.
    for (int i = tid * 4; i < WORDS_PER_IMG; i += 1024 * 4) {
        async_g2l_b128(&S[i],  &strongBM[base + i]);
        async_g2l_b128(&Wk[i], &weakBM[base + i]);
    }
    if (tid == 0) flag[0] = 0u;
    wait_asynccnt0();
    __syncthreads();

    for (int it = 0; it < 256; ++it) {
        unsigned changed = 0u;
        for (int i = tid; i < WORDS_PER_IMG; i += 1024) {
            int row = i >> 5, wc = i & 31;
            unsigned s = S[i];
            unsigned w = Wk[i];
            if ((w & ~s) == 0u) continue;   // nothing can grow here
            auto hor = [&](int rr) -> unsigned {
                if (rr < 0 || rr > 1023) return 0u;
                int j = (rr << 5) + wc;
                unsigned xv = S[j];
                unsigned lv = wc ? S[j - 1] : 0u;
                unsigned rv = (wc < 31) ? S[j + 1] : 0u;
                return xv | (xv << 1) | (lv >> 31) | (xv >> 1) | (rv << 31);
            };
            unsigned dil = hor(row - 1) | hor(row) | hor(row + 1);
            unsigned ns = s | (dil & w);
            if (ns != s) { S[i] = ns; changed = 1u; }
        }
        if (changed) atomicOr(flag, 1u);
        __syncthreads();
        unsigned f = flag[0];
        __syncthreads();
        if (!f) break;
        if (tid == 0) flag[0] = 0u;
        __syncthreads();
    }
    __syncthreads();

    // Async bulk writeback of the final edge bitmap.
    for (int i = tid * 4; i < WORDS_PER_IMG; i += 1024 * 4)
        async_l2g_b128(&strongBM[base + i], &S[i]);
    wait_asynccnt0();
}

// ---------------------------------------------------------------------------
// Loss: sum_n mean_{h,w} |edge*mask - y*mask| = (1/HW) * grand_sum.
// Deterministic two-stage tree reduction (no float atomics).
// ---------------------------------------------------------------------------
__global__ __launch_bounds__(256) void loss_partial(const unsigned* __restrict__ edges,
                                                    const float* __restrict__ y,
                                                    const float* __restrict__ mask,
                                                    float* __restrict__ partial) {
    __shared__ float red[256];
    const int tid = threadIdx.x;
    float s = 0.0f;
#pragma unroll
    for (int k = 0; k < 16; ++k) {
        int p = blockIdx.x * 4096 + k * 256 + tid;
        if (k + 1 < 16)
            __builtin_prefetch(&y[p + 256], 0, 1);   // global_prefetch_b8
        unsigned wbits = edges[p >> 5];
        float e = ((wbits >> (p & 31)) & 1u) ? 1.0f : 0.0f;
        float mk = mask[p & (HW - 1)];
        s += fabsf(e * mk - y[p] * mk);
    }
    red[tid] = s;
    __syncthreads();
    for (int off = 128; off; off >>= 1) {
        if (tid < off) red[tid] += red[tid + off];
        __syncthreads();
    }
    if (tid == 0) partial[blockIdx.x] = red[0];
}

__global__ __launch_bounds__(256) void loss_final(const float* __restrict__ partial,
                                                  float* __restrict__ out) {
    __shared__ float red[256];
    const int tid = threadIdx.x;
    float s = 0.0f;
    for (int i = tid; i < 4096; i += 256) s += partial[i];
    red[tid] = s;
    __syncthreads();
    for (int off = 128; off; off >>= 1) {
        if (tid < off) red[tid] += red[tid + off];
        __syncthreads();
    }
    if (tid == 0) out[0] = red[0] * (1.0f / (float)HW);
}

// ---------------------------------------------------------------------------
extern "C" void kernel_launch(void* const* d_in, const int* in_sizes, int n_in,
                              void* d_out, int out_size, void* d_ws, size_t ws_size,
                              hipStream_t stream) {
    const float* x    = (const float*)d_in[0];   // (16,1,1024,1024)
    const float* y    = (const float*)d_in[1];   // (16,1,1024,1024)
    const float* mask = (const float*)d_in[2];   // (1024,1024)
    float* out = (float*)d_out;

    char* ws = (char*)d_ws;
    float*    bufA     = (float*)ws;                          // 67108864 B
    float*    bufB     = (float*)(ws + 67108864);             // 67108864 B
    unsigned* strongBM = (unsigned*)(ws + 134217728);         //  2097152 B
    unsigned* weakBM   = (unsigned*)(ws + 136314880);         //  2097152 B
    float*    partial  = (float*)(ws + 138412032);            //    16384 B

    // 65536 16x16 tiles total, 8 waves (tiles) per 256-thread block
    blur_h_wmma<<<8192, 256, 0, stream>>>(x, bufA);
    blur_v_wmma<<<8192, 256, 0, stream>>>(bufA, bufB);        // bufB = smoothed
    mag_k<<<NHW / 256, 256, 0, stream>>>(bufB, bufA);         // bufA = |grad|
    nms_k<<<NHW / 256, 256, 0, stream>>>(bufB, bufA, strongBM, weakBM);
    hyst_k<<<NIMG, 1024, (2 * WORDS_PER_IMG + 1) * sizeof(unsigned), stream>>>(
        strongBM, weakBM);
    loss_partial<<<4096, 256, 0, stream>>>(strongBM, y, mask, partial);
    loss_final<<<1, 256, 0, stream>>>(partial, out);
}
